// MultiHeadAttention_3582002725198
// MI455X (gfx1250) — matrix operand order repaired and checked by reading, not compiled, not run on hardware
//
#include <hip/hip_runtime.h>
#include <hip/hip_bf16.h>

// Problem constants (match reference)
#define D_MODEL 1024
#define NHEADS  16
#define DK      64
#define BATCH   4
#define SEQ     2048
#define MROWS   (BATCH * SEQ)   // 8192 rows in all flattened GEMMs

typedef __attribute__((ext_vector_type(16))) _Float16 v16h;
typedef __attribute__((ext_vector_type(8)))  _Float16 v8h;
typedef __attribute__((ext_vector_type(8)))  float    v8f;

// TDM descriptor groups (6-arg tensor_load_to_lds builtin on this toolchain)
typedef __attribute__((ext_vector_type(4))) unsigned int u32x4;
typedef __attribute__((ext_vector_type(8))) int          i32x8;
typedef __attribute__((ext_vector_type(4))) int          i32x4;

union H16 { v16h v; v8h h8[2]; _Float16 h[16]; };
union H8  { v8h  v; _Float16 h[8]; };

// async LDS-DMA builtin takes <4 x i32>* with explicit address spaces
#define GLB128(p) ((__attribute__((address_space(1))) i32x4*)(void*)(p))
#define LDS128(p) ((__attribute__((address_space(3))) i32x4*)(void*)(p))

// ---------------------------------------------------------------------------
// f32 -> f16 conversion (one-time, bandwidth-trivial)
// ---------------------------------------------------------------------------
__global__ void cvt_f32_f16(const float* __restrict__ in,
                            _Float16* __restrict__ out, int n) {
  int i = blockIdx.x * blockDim.x + threadIdx.x;
  int stride = gridDim.x * blockDim.x;
  for (; i < n; i += stride) out[i] = (_Float16)in[i];
}

// ---------------------------------------------------------------------------
// WMMA GEMM: C[M,N] = A[M,K] * B[K,N] + bias
//   OUT_MODE 0: f16 output scattered into [B,H,S,dk] head layout (projections)
//   OUT_MODE 1: f32 output, flat [M,N] (final output projection -> d_out)
// Block: 256 threads (8 waves). Tile: BM=128 x BN=64, BK=32.
// A tile is staged with global_load_async_to_lds_b128 (ASYNCcnt path).
// ---------------------------------------------------------------------------
template <int OUT_MODE>
__global__ __launch_bounds__(256)
void gemm_wmma(const _Float16* __restrict__ A, const _Float16* __restrict__ Bw,
               const float* __restrict__ bias, void* __restrict__ outp) {
  constexpr int M = MROWS, N = D_MODEL, K = D_MODEL;
  constexpr int BM = 128, BN = 64, BK = 32;

  __shared__ __align__(16) _Float16 Ash[BM][BK + 8];   // row-major A tile
  __shared__ __align__(16) _Float16 Bt[BN][BK + 8];    // transposed B tile

  const int tid  = threadIdx.x;
  const int wave = tid >> 5;
  const int lane = tid & 31;
  const int l16  = lane & 15;
  const int mbase = blockIdx.x * BM;
  const int nbase = blockIdx.y * BN;

  const v8f vz = {0.f, 0.f, 0.f, 0.f, 0.f, 0.f, 0.f, 0.f};
  v8f acc[4];
#pragma unroll
  for (int t = 0; t < 4; ++t) acc[t] = vz;

  const int arow  = tid >> 1;          // 0..127
  const int ahalf = tid & 1;           // which 16-f16 half of the 32-wide row
  const int bkrow = tid >> 3;          // 0..31
  const int bnc   = (tid & 7) << 3;    // 0,8,..,56

  for (int kk = 0; kk < K; kk += BK) {
    // --- stage A tile: async LDS-DMA, no VGPR round trip ---
    {
      const _Float16* src = A + (size_t)(mbase + arow) * K + kk + ahalf * 16;
      __builtin_amdgcn_global_load_async_to_lds_b128(
          GLB128(src), LDS128(&Ash[arow][ahalf * 16]), 0, 0);
      __builtin_amdgcn_global_load_async_to_lds_b128(
          GLB128(src + 8), LDS128(&Ash[arow][ahalf * 16 + 8]), 0, 0);
    }
    // --- stage B tile, transposed so B-operand reads are contiguous ---
    {
      const _Float16* src = Bw + (size_t)(kk + bkrow) * N + nbase + bnc;
      H8 b;
      b.v = *(const v8h*)src;
#pragma unroll
      for (int j = 0; j < 8; ++j) Bt[bnc + j][bkrow] = b.h[j];
    }
    if (kk + BK < K) {   // global_prefetch_b8 next K-step
      __builtin_prefetch(A + (size_t)(mbase + arow) * K + kk + BK + ahalf * 16, 0, 0);
      __builtin_prefetch(Bw + (size_t)(kk + BK + bkrow) * N + nbase + bnc, 0, 0);
    }
    __builtin_amdgcn_s_wait_asynccnt(0);
    __syncthreads();

    // A operand: 16-bit A layout. lanes 0-15 hold K {0..7,16..23} of row l16;
    // lanes 16-31 hold K {8..15,24..31}.
    const int mrow = wave * 16 + l16;
    const int k0   = (lane < 16) ? 0 : 8;
    H16 a;
    a.h8[0] = *(const v8h*)&Ash[mrow][k0];
    a.h8[1] = *(const v8h*)&Ash[mrow][k0 + 16];

#pragma unroll
    for (int t = 0; t < 4; ++t) {
      const int col = t * 16 + l16;
      const int kb  = (lane < 16) ? 0 : 8;
      H16 b;
      b.h8[0] = *(const v8h*)&Bt[col][kb];
      b.h8[1] = *(const v8h*)&Bt[col][kb + 16];
      acc[t] = __builtin_amdgcn_wmma_f32_16x16x32_f16(
          false, a.v, false, b.v, (short)0, acc[t], false, false);
    }
    __syncthreads();
  }

  // --- epilogue: bias + store ---
  const int rowOff = (lane < 16) ? 0 : 8;   // C layout: lanes16-31 -> M=8+r
#pragma unroll
  for (int t = 0; t < 4; ++t) {
    const int n = nbase + t * 16 + l16;
    const float bv = bias[n];
#pragma unroll
    for (int r = 0; r < 8; ++r) {
      const int m = mbase + wave * 16 + rowOff + r;
      const float val = acc[t][r] + bv;
      if (OUT_MODE == 0) {
        const int b = m >> 11, s = m & (SEQ - 1);     // SEQ = 2048
        const int h = n >> 6,  d = n & (DK - 1);
        ((_Float16*)outp)[((size_t)(b * NHEADS + h) * SEQ + s) * DK + d] =
            (_Float16)val;
      } else {
        ((float*)outp)[(size_t)m * N + n] = val;
      }
    }
  }
}

// ---------------------------------------------------------------------------
// Flash attention over [B,H,S,dk] f16 projections.
// Grid: (SEQ/64 q-tiles, BATCH*NHEADS). Block: 128 threads = 4 waves.
// K block is staged by the Tensor Data Mover (tensor_load_to_lds, TENSORcnt);
// the D# descriptor's LDS padding (4 DW per 32 DW) reproduces the [64][72]
// bank-conflict-free layout. V is transposed manually (TDM cannot transpose).
// ---------------------------------------------------------------------------
__global__ __launch_bounds__(128)
void attn_flash(const _Float16* __restrict__ Qp, const _Float16* __restrict__ Kp,
                const _Float16* __restrict__ Vp, _Float16* __restrict__ Ctx) {
  __shared__ __align__(16) _Float16 Ksh[64][72];      // [key][dk]
  __shared__ __align__(16) _Float16 Vt[64][72];       // [d][key] (transposed)
  __shared__ __align__(16) _Float16 Psh[4][16][72];   // per-wave P relayout

  const int tid  = threadIdx.x;
  const int wave = tid >> 5;
  const int lane = tid & 31;
  const int l16  = lane & 15;
  const int bh    = blockIdx.y;
  const int qbase = blockIdx.x * 64;
  const size_t base = (size_t)bh * SEQ * DK;

  // Wave-uniform scalar predicate: only wave 0 drives the TDM (TDM ignores
  // EXEC, so this must be a scalar branch, not an EXEC-masked region).
  const int waveScalar = __builtin_amdgcn_readfirstlane(wave);

  // TDM descriptor (constant parts). Group1 bitfields per ISA 8.4:
  //   data_size=1 (2B)            -> bits 17:16
  //   pad_enable=1                -> bit 20
  //   pad_interval=4 (32 DWORDs)  -> bits 24:22   (one 64-f16 row = 32 DW)
  //   pad_amount=3 (4 DWORDs)     -> bits 31:25   (8 f16 row padding)
  //   tensor_dim0=64, tensor_dim1=SEQ, tile 64x64, dim0_stride=64
  const unsigned ldsK =
      (unsigned)(unsigned long long)(__attribute__((address_space(3))) _Float16*)&Ksh[0][0];
  const i32x8 g1 = {
      (1 << 16) | (1 << 20) | (4 << 22) | (3 << 25),  // w0
      (DK & 0xffff) << 16,                            // w1: tensor_dim0 lo16
      (SEQ & 0xffff) << 16,                           // w2: tensor_dim1 lo16
      (DK & 0xffff) << 16,                            // w3: tile_dim0
      64,                                             // w4: tile_dim1
      DK,                                             // w5: tensor_dim0_stride
      0, 0};
  const i32x4 gz4 = {0, 0, 0, 0};
  const i32x8 gz8 = {0, 0, 0, 0, 0, 0, 0, 0};

  // Q A-operands (dk = 64 -> two 16x32 chunks), A VGPR layout per lane-half.
  const int k0 = (lane < 16) ? 0 : 8;
  const _Float16* qrow = Qp + base + (size_t)(qbase + wave * 16 + l16) * DK;
  H16 aq0, aq1;
  aq0.h8[0] = *(const v8h*)(qrow + k0);
  aq0.h8[1] = *(const v8h*)(qrow + k0 + 16);
  aq1.h8[0] = *(const v8h*)(qrow + 32 + k0);
  aq1.h8[1] = *(const v8h*)(qrow + 48 + k0);

  float rm[8], rl[8];
  const v8f vz = {0.f, 0.f, 0.f, 0.f, 0.f, 0.f, 0.f, 0.f};
  v8f o[4];
#pragma unroll
  for (int r = 0; r < 8; ++r) { rm[r] = -__builtin_inff(); rl[r] = 0.f; }
#pragma unroll
  for (int t = 0; t < 4; ++t) o[t] = vz;

  const int srow  = tid >> 1;   // 0..63
  const int shalf = tid & 1;    // which 32-wide half of dk

  for (int kb = 0; kb < SEQ; kb += 64) {
    // --- stage K block via Tensor Data Mover (wave 0 only) ---
    if (waveScalar == 0) {
      const unsigned long long ga =
          (unsigned long long)(Kp + base + (size_t)kb * DK);
      const u32x4 g0 = {
          1u,                                           // count=1, user D#
          ldsK,                                         // lds_addr
          (unsigned)ga,                                 // global_addr[31:0]
          (unsigned)((ga >> 32) & 0x1FFFFFFull) | 0x80000000u};  // addr|type=2
      __builtin_amdgcn_tensor_load_to_lds(g0, g1, gz4, gz4, gz8, 0);
    }
    // --- stage V block transposed: Vt[d][key] (all waves, overlaps TDM) ---
    {
      const _Float16* src = Vp + base + (size_t)(kb + srow) * DK + shalf * 32;
#pragma unroll
      for (int j = 0; j < 32; ++j) Vt[shalf * 32 + j][srow] = src[j];
    }
    if (waveScalar == 0) __builtin_amdgcn_s_wait_tensorcnt(0);
    __syncthreads();

    // --- scores: S = Q * K^T (4 col-tiles x 2 K-chunks) ---
    v8f st[4];
    const int kbb = (lane < 16) ? 0 : 8;
#pragma unroll
    for (int t = 0; t < 4; ++t) {
      const int col = t * 16 + l16;   // key column this lane holds
      H16 b0, b1;
      b0.h8[0] = *(const v8h*)&Ksh[col][kbb];
      b0.h8[1] = *(const v8h*)&Ksh[col][kbb + 16];
      b1.h8[0] = *(const v8h*)&Ksh[col][32 + kbb];
      b1.h8[1] = *(const v8h*)&Ksh[col][32 + kbb + 16];
      v8f s = __builtin_amdgcn_wmma_f32_16x16x32_f16(
          false, aq0.v, false, b0.v, (short)0, vz, false, false);
      s = __builtin_amdgcn_wmma_f32_16x16x32_f16(
          false, aq1.v, false, b1.v, (short)0, s, false, false);
      st[t] = s;
    }

    // --- scale ---
#pragma unroll
    for (int t = 0; t < 4; ++t)
#pragma unroll
      for (int r = 0; r < 8; ++r) st[t][r] *= 0.125f;   // 1/sqrt(64)

    // --- online softmax (row reductions across 16-lane halves) ---
    float corr[8];
#pragma unroll
    for (int r = 0; r < 8; ++r) {
      float m = fmaxf(fmaxf(st[0][r], st[1][r]), fmaxf(st[2][r], st[3][r]));
#pragma unroll
      for (int off = 8; off; off >>= 1) m = fmaxf(m, __shfl_xor(m, off, 32));
      const float mnew = fmaxf(rm[r], m);
      corr[r] = __expf(rm[r] - mnew);
      rm[r] = mnew;
    }
    float psum[8];
#pragma unroll
    for (int r = 0; r < 8; ++r) psum[r] = 0.f;
#pragma unroll
    for (int t = 0; t < 4; ++t)
#pragma unroll
      for (int r = 0; r < 8; ++r) {
        const float p = __expf(st[t][r] - rm[r]);
        st[t][r] = p;
        psum[r] += p;
      }
#pragma unroll
    for (int r = 0; r < 8; ++r) {
      float s = psum[r];
#pragma unroll
      for (int off = 8; off; off >>= 1) s += __shfl_xor(s, off, 32);
      rl[r] = rl[r] * corr[r] + s;
    }
#pragma unroll
    for (int t = 0; t < 4; ++t)
#pragma unroll
      for (int r = 0; r < 8; ++r) o[t][r] *= corr[r];

    // --- relayout P: C layout -> A layout via per-wave LDS tile ---
    const int prow = (lane < 16) ? 0 : 8;
#pragma unroll
    for (int t = 0; t < 4; ++t)
#pragma unroll
      for (int r = 0; r < 8; ++r)
        Psh[wave][prow + r][t * 16 + l16] = (_Float16)st[t][r];
    __syncthreads();

    H16 ap0, ap1;
    ap0.h8[0] = *(const v8h*)&Psh[wave][l16][k0];
    ap0.h8[1] = *(const v8h*)&Psh[wave][l16][k0 + 16];
    ap1.h8[0] = *(const v8h*)&Psh[wave][l16][32 + k0];
    ap1.h8[1] = *(const v8h*)&Psh[wave][l16][48 + k0];

    // --- O += P * V ---
#pragma unroll
    for (int t = 0; t < 4; ++t) {
      const int col = t * 16 + l16;   // output d column
      H16 b0, b1;
      b0.h8[0] = *(const v8h*)&Vt[col][kbb];
      b0.h8[1] = *(const v8h*)&Vt[col][kbb + 16];
      b1.h8[0] = *(const v8h*)&Vt[col][32 + kbb];
      b1.h8[1] = *(const v8h*)&Vt[col][32 + kbb + 16];
      o[t] = __builtin_amdgcn_wmma_f32_16x16x32_f16(
          false, ap0.v, false, b0.v, (short)0, o[t], false, false);
      o[t] = __builtin_amdgcn_wmma_f32_16x16x32_f16(
          false, ap1.v, false, b1.v, (short)0, o[t], false, false);
    }
    __syncthreads();
  }

  // --- normalize and store ctx as [B,S,D] f16 ---
  const int b = bh >> 4, h = bh & 15;
#pragma unroll
  for (int t = 0; t < 4; ++t)
#pragma unroll
    for (int r = 0; r < 8; ++r) {
      const int q = qbase + wave * 16 + ((lane < 16) ? 0 : 8) + r;
      const float val = o[t][r] / rl[r];
      Ctx[(size_t)(b * SEQ + q) * D_MODEL + h * DK + t * 16 + l16] =
          (_Float16)val;
    }
}

// ---------------------------------------------------------------------------
// Host launcher
// ---------------------------------------------------------------------------
extern "C" void kernel_launch(void* const* d_in, const int* in_sizes, int n_in,
                              void* d_out, int out_size, void* d_ws,
                              size_t ws_size, hipStream_t stream) {
  const float* Q  = (const float*)d_in[0];
  const float* K  = (const float*)d_in[1];
  const float* V  = (const float*)d_in[2];
  const float* Wq = (const float*)d_in[3];
  const float* bq = (const float*)d_in[4];
  const float* Wk = (const float*)d_in[5];
  const float* bk = (const float*)d_in[6];
  const float* Wv = (const float*)d_in[7];
  const float* bv = (const float*)d_in[8];
  const float* Wo = (const float*)d_in[9];
  const float* bo = (const float*)d_in[10];

  const size_t NX = (size_t)MROWS * D_MODEL;      // 8,388,608 elements
  const size_t NW = (size_t)D_MODEL * D_MODEL;    // 1,048,576 elements

  _Float16* ws  = (_Float16*)d_ws;
  _Float16* Xq  = ws;  ws += NX;
  _Float16* Xk  = ws;  ws += NX;
  _Float16* Xv  = ws;  ws += NX;
  _Float16* Wqf = ws;  ws += NW;
  _Float16* Wkf = ws;  ws += NW;
  _Float16* Wvf = ws;  ws += NW;
  _Float16* Wof = ws;  ws += NW;
  _Float16* Qp  = ws;  ws += NX;   // [B,H,S,dk]
  _Float16* Kp  = ws;  ws += NX;
  _Float16* Vp  = ws;  ws += NX;
  _Float16* Ctx = ws;  ws += NX;   // [B,S,D]

  const int cvtB = 4096, cvtT = 256;
  cvt_f32_f16<<<cvtB, cvtT, 0, stream>>>(Q,  Xq,  (int)NX);
  cvt_f32_f16<<<cvtB, cvtT, 0, stream>>>(K,  Xk,  (int)NX);
  cvt_f32_f16<<<cvtB, cvtT, 0, stream>>>(V,  Xv,  (int)NX);
  cvt_f32_f16<<<1024, cvtT, 0, stream>>>(Wq, Wqf, (int)NW);
  cvt_f32_f16<<<1024, cvtT, 0, stream>>>(Wk, Wkf, (int)NW);
  cvt_f32_f16<<<1024, cvtT, 0, stream>>>(Wv, Wvf, (int)NW);
  cvt_f32_f16<<<1024, cvtT, 0, stream>>>(Wo, Wof, (int)NW);

  dim3 gg(MROWS / 128, D_MODEL / 64);   // 64 x 16 blocks
  gemm_wmma<0><<<gg, 256, 0, stream>>>(Xq, Wqf, bq, Qp);
  gemm_wmma<0><<<gg, 256, 0, stream>>>(Xk, Wkf, bk, Kp);
  gemm_wmma<0><<<gg, 256, 0, stream>>>(Xv, Wvf, bv, Vp);

  attn_flash<<<dim3(SEQ / 64, BATCH * NHEADS), 128, 0, stream>>>(Qp, Kp, Vp, Ctx);

  gemm_wmma<1><<<gg, 256, 0, stream>>>(Ctx, Wof, bo, d_out);
}
